// EncoderLayer_42107859370756
// MI455X (gfx1250) — matrix operand order repaired and checked by reading, not compiled, not run on hardware
//
#include <hip/hip_runtime.h>

#define B_  2
#define S_  2048
#define H_  1024
#define NH_ 16
#define HD_ 64

typedef unsigned short ushort8v __attribute__((ext_vector_type(8)));
typedef __bf16 bf16x16 __attribute__((ext_vector_type(16)));
typedef __bf16 bf16x2 __attribute__((ext_vector_type(2)));
typedef float floatx8 __attribute__((ext_vector_type(8)));
typedef int async_v4i __attribute__((vector_size(16)));   // matches builtin param type

union Frag { ushort8v h[2]; bf16x16 b; };
union F8   { floatx8 v; float f[8]; };

#if defined(__has_builtin)
#if __has_builtin(__builtin_amdgcn_global_load_async_to_lds_b128) && \
    __has_builtin(__builtin_amdgcn_s_wait_asynccnt)
#define USE_ASYNC_LDS 1
#endif
#if __has_builtin(__builtin_amdgcn_cvt_pk_bf16_f32)
#define USE_CVT_PK 1
#endif
#endif
#ifndef USE_ASYNC_LDS
#define USE_ASYNC_LDS 0
#endif
#ifndef USE_CVT_PK
#define USE_CVT_PK 0
#endif

__device__ __forceinline__ unsigned int pack_bf16x2(float lo, float hi) {
#if USE_CVT_PK
  union { bf16x2 v; unsigned int u; } x;
  x.v = __builtin_amdgcn_cvt_pk_bf16_f32(lo, hi);   // v_cvt_pk_bf16_f32 (RNE)
  return x.u;
#else
  // native fptrunc f32->bf16: backend picks the hardware convert (RNE)
  union { bf16x2 v; unsigned int u; } x;
  bf16x2 t = {(__bf16)lo, (__bf16)hi};
  x.v = t;
  return x.u;
#endif
}

__device__ __forceinline__ unsigned short f32_to_bf16(float x) {
  union { __bf16 b; unsigned short s; } t;
  t.b = (__bf16)x;
  return t.s;
}

// Load a 16-element bf16 fragment from LDS as two 16-byte chunks.
__device__ __forceinline__ bf16x16 ld_frag(const unsigned short* p, int o0, int o1) {
  Frag f;
  f.h[0] = *(const ushort8v*)(p + o0);
  f.h[1] = *(const ushort8v*)(p + o1);
  return f.b;
}

__device__ __forceinline__ floatx8 wmma_bf16(bf16x16 a, bf16x16 b, floatx8 c) {
  return __builtin_amdgcn_wmma_f32_16x16x32_bf16(false, a, false, b, (short)0, c, false, false);
}

// ---------------------------------------------------------------------------
// C[M,N] = A[M,K] * W[K,N] + bias.  M=4096, N=K=1024.
// Block: 256 threads = 8 waves (2 x 4). Block tile 128x128, K step 32.
// Wave tile: 64(M) x 32(N) = 4 x 2 WMMA tiles.
// ---------------------------------------------------------------------------
__global__ __launch_bounds__(256) void gemm_bias_kernel(
    const float* __restrict__ A, const float* __restrict__ W,
    const float* __restrict__ bias, float* __restrict__ C) {
  const int N = 1024, K = 1024;
  __shared__ __align__(16) unsigned short As[128 * 32];   // A tile (row-major), bf16 bits
  __shared__ __align__(16) unsigned short Bt[128 * 32];   // W tile transposed: Bt[n][k]
  const int tid = threadIdx.x;
  const int lane = tid & 31, wave = tid >> 5;
  const int lane15 = lane & 15, hl = lane >> 4;
  const int wm = wave >> 2, wn = wave & 3;
  const int m0 = blockIdx.y * 128, n0 = blockIdx.x * 128;

  F8 acc[4][2];
  for (int mi = 0; mi < 4; ++mi)
    for (int ni = 0; ni < 2; ++ni)
      for (int v = 0; v < 8; ++v) acc[mi][ni].f[v] = 0.f;

  for (int k0 = 0; k0 < K; k0 += 32) {
    // A tile: 128x32 fp32 -> bf16, vectorized (float4 in, b64 out)
#pragma unroll
    for (int i = 0; i < 4; ++i) {
      int idx4 = (i * 256 + tid) * 4;
      int r = idx4 >> 5, c = idx4 & 31;
      float4 f = *(const float4*)(A + (size_t)(m0 + r) * K + k0 + c);
      uint2 p;
      p.x = pack_bf16x2(f.x, f.y);
      p.y = pack_bf16x2(f.z, f.w);
      *(uint2*)&As[r * 32 + c] = p;
    }
    // W tile: 32x128 fp32, stored transposed Bt[n][k]; 4 k's per thread packed
#pragma unroll
    for (int i = 0; i < 4; ++i) {
      int slot = i * 256 + tid;
      int n = slot & 127, kk = (slot >> 7) * 4;
      const float* wp = W + (size_t)(k0 + kk) * N + n0 + n;
      uint2 p;
      p.x = pack_bf16x2(wp[0], wp[N]);
      p.y = pack_bf16x2(wp[2 * N], wp[3 * N]);
      *(uint2*)&Bt[n * 32 + kk] = p;
    }
    // prefetch next K tile (one cacheline per thread covers both tiles)
    if (k0 + 32 < K) {
      if (tid < 128) {
        __builtin_prefetch(A + (size_t)(m0 + tid) * K + k0 + 32, 0, 1);
      } else {
        int t2 = tid - 128;
        __builtin_prefetch(W + (size_t)(k0 + 32 + (t2 >> 2)) * N + n0 + (t2 & 3) * 32, 0, 1);
      }
    }
    __syncthreads();

    bf16x16 af[4];
#pragma unroll
    for (int mi = 0; mi < 4; ++mi)
      af[mi] = ld_frag(&As[(wm * 64 + mi * 16 + lane15) * 32], 8 * hl, 16 + 8 * hl);
#pragma unroll
    for (int ni = 0; ni < 2; ++ni) {
      bf16x16 bfr = ld_frag(&Bt[(wn * 32 + ni * 16 + lane15) * 32], 8 * hl, 16 + 8 * hl);
#pragma unroll
      for (int mi = 0; mi < 4; ++mi)
        acc[mi][ni].v = wmma_bf16(af[mi], bfr, acc[mi][ni].v);
    }
    __syncthreads();
  }

#pragma unroll
  for (int ni = 0; ni < 2; ++ni) {
    int n = n0 + wn * 32 + ni * 16 + lane15;
    float bv = bias[n];
#pragma unroll
    for (int mi = 0; mi < 4; ++mi)
#pragma unroll
      for (int v = 0; v < 8; ++v) {
        int m = m0 + wm * 64 + mi * 16 + 8 * hl + v;
        C[m * N + n] = acc[mi][ni].f[v] + bv;
      }
  }
}

// ---------------------------------------------------------------------------
// Flash-style attention. One workgroup = (batch, head, 64-row query tile).
// 128 threads = 4 waves, each wave owns 16 query rows. Key blocks of 64.
// ---------------------------------------------------------------------------
__global__ __launch_bounds__(128) void attn_kernel(
    const float* __restrict__ Q, const float* __restrict__ Kp,
    const float* __restrict__ Vp, const float* __restrict__ Mask,
    float* __restrict__ Ctx) {
  __shared__ __align__(16) unsigned short Qs[64 * 64];   // scaled Q, row-major
  __shared__ __align__(16) unsigned short Ks[64 * 64];   // K rows (= B cols of Q*K^T)
  __shared__ __align__(16) unsigned short Vst[64 * 64];  // V transposed: Vst[d][kk]
  __shared__ __align__(16) unsigned short Ps[64 * 64];   // probabilities, bf16
  __shared__ __align__(16) float Ms[64 * 64];            // mask tile
  const int tid = threadIdx.x;
  const int lane = tid & 31, w = tid >> 5;
  const int lane15 = lane & 15, hl = lane >> 4;
  const int q0 = blockIdx.x * 64;
  const int head = blockIdx.y;
  const int b = blockIdx.z;
  const float scale = 0.125f;  // 1/sqrt(64)

#pragma unroll
  for (int i = 0; i < 8; ++i) {
    int idx4 = (i * 128 + tid) * 4;
    int r = idx4 >> 6, c = idx4 & 63;
    float4 f = *(const float4*)(Q + (size_t)(b * S_ + q0 + r) * H_ + head * HD_ + c);
    uint2 p;
    p.x = pack_bf16x2(scale * f.x, scale * f.y);
    p.y = pack_bf16x2(scale * f.z, scale * f.w);
    *(uint2*)&Qs[r * 64 + c] = p;
  }

  F8 O[4];
  float m_run[8], l_run[8];
  for (int t = 0; t < 4; ++t)
    for (int v = 0; v < 8; ++v) O[t].f[v] = 0.f;
  for (int v = 0; v < 8; ++v) { m_run[v] = -1e30f; l_run[v] = 0.f; }

  for (int j0 = 0; j0 < S_; j0 += 64) {
    // K tile (row-major) — vectorized convert-to-bf16
#pragma unroll
    for (int i = 0; i < 8; ++i) {
      int idx4 = (i * 128 + tid) * 4;
      int r = idx4 >> 6, c = idx4 & 63;
      float4 f = *(const float4*)(Kp + (size_t)(b * S_ + j0 + r) * H_ + head * HD_ + c);
      uint2 p;
      p.x = pack_bf16x2(f.x, f.y);
      p.y = pack_bf16x2(f.z, f.w);
      *(uint2*)&Ks[r * 64 + c] = p;
    }
    // V tile transposed: Vst[d][kk]; 4 kk's per thread packed along kk
#pragma unroll
    for (int i = 0; i < 8; ++i) {
      int slot = i * 128 + tid;
      int d = slot & 63, kk = (slot >> 6) * 4;
      const float* vp = Vp + (size_t)(b * S_ + j0 + kk) * H_ + head * HD_ + d;
      uint2 p;
      p.x = pack_bf16x2(vp[0], vp[H_]);
      p.y = pack_bf16x2(vp[2 * H_], vp[3 * H_]);
      *(uint2*)&Vst[d * 64 + kk] = p;
    }
    // Mask tile: pure fp32 copy -> async global->LDS path when available
#if USE_ASYNC_LDS
#pragma unroll
    for (int i = 0; i < 8; ++i) {
      int idx4 = (i * 128 + tid) * 4;
      int r = idx4 >> 6, c = idx4 & 63;
      __builtin_amdgcn_global_load_async_to_lds_b128(
          (__attribute__((address_space(1))) async_v4i*)(Mask + (size_t)(b * S_ + q0 + r) * S_ + j0 + c),
          (__attribute__((address_space(3))) async_v4i*)(&Ms[r * 64 + c]), 0, 0);
    }
    __builtin_amdgcn_s_wait_asynccnt(0);
#else
#pragma unroll
    for (int i = 0; i < 8; ++i) {
      int idx4 = (i * 128 + tid) * 4;
      int r = idx4 >> 6, c = idx4 & 63;
      *(float4*)&Ms[r * 64 + c] =
          *(const float4*)(Mask + (size_t)(b * S_ + q0 + r) * S_ + j0 + c);
    }
#endif
    __syncthreads();

    // scores: S = (Q*scale) @ K^T  -> 4 tiles of 16x16 per wave
    F8 sc[4];
    for (int t = 0; t < 4; ++t)
      for (int v = 0; v < 8; ++v) sc[t].f[v] = 0.f;
#pragma unroll
    for (int kk0 = 0; kk0 < 64; kk0 += 32) {
      bf16x16 a = ld_frag(&Qs[(w * 16 + lane15) * 64], kk0 + 8 * hl, kk0 + 16 + 8 * hl);
#pragma unroll
      for (int t = 0; t < 4; ++t) {
        bf16x16 bb = ld_frag(&Ks[(t * 16 + lane15) * 64], kk0 + 8 * hl, kk0 + 16 + 8 * hl);
        sc[t].v = wmma_bf16(a, bb, sc[t].v);
      }
    }

    // online softmax (row = v + 8*hl within wave tile; 16 N-values per 16-lane half)
#pragma unroll
    for (int v = 0; v < 8; ++v) {
      int rq = w * 16 + 8 * hl + v;
      float s[4];
      float mx = -1e30f;
#pragma unroll
      for (int t = 0; t < 4; ++t) {
        s[t] = sc[t].f[v] - 10000.f * Ms[rq * 64 + t * 16 + lane15];
        mx = fmaxf(mx, s[t]);
      }
#pragma unroll
      for (int off = 1; off < 16; off <<= 1) mx = fmaxf(mx, __shfl_xor(mx, off, 32));
      float mnew = fmaxf(m_run[v], mx);
      float p[4];
      float rs = 0.f;
#pragma unroll
      for (int t = 0; t < 4; ++t) { p[t] = __expf(s[t] - mnew); rs += p[t]; }
#pragma unroll
      for (int off = 1; off < 16; off <<= 1) rs += __shfl_xor(rs, off, 32);
      float fac = __expf(m_run[v] - mnew);
      l_run[v] = l_run[v] * fac + rs;
      m_run[v] = mnew;
#pragma unroll
      for (int t = 0; t < 4; ++t) {
        O[t].f[v] *= fac;
        Ps[rq * 64 + t * 16 + lane15] = f32_to_bf16(p[t]);
      }
    }

    // O += P @ V  (wave reads only its own 16 Ps rows: no barrier needed)
#pragma unroll
    for (int kk0 = 0; kk0 < 64; kk0 += 32) {
      bf16x16 a = ld_frag(&Ps[(w * 16 + lane15) * 64], kk0 + 8 * hl, kk0 + 16 + 8 * hl);
#pragma unroll
      for (int t2 = 0; t2 < 4; ++t2) {
        bf16x16 bb = ld_frag(&Vst[(t2 * 16 + lane15) * 64], kk0 + 8 * hl, kk0 + 16 + 8 * hl);
        O[t2].v = wmma_bf16(a, bb, O[t2].v);
      }
    }
    __syncthreads();
  }

#pragma unroll
  for (int t2 = 0; t2 < 4; ++t2) {
    int d = t2 * 16 + lane15;
#pragma unroll
    for (int v = 0; v < 8; ++v) {
      int rq = w * 16 + 8 * hl + v;
      Ctx[(b * S_ + q0 + rq) * H_ + head * HD_ + d] = O[t2].f[v] / l_run[v];
    }
  }
}

// ---------------------------------------------------------------------------
// out = LayerNorm(X + R) * gamma + beta ; one row (H=1024) per workgroup
// ---------------------------------------------------------------------------
__global__ __launch_bounds__(256) void ln_kernel(
    const float* __restrict__ X, const float* __restrict__ R,
    const float* __restrict__ gamma, const float* __restrict__ beta,
    float* __restrict__ Out) {
  __shared__ float red[256];
  const int row = blockIdx.x, t = threadIdx.x;
  float vals[4];
  float s = 0.f;
#pragma unroll
  for (int i = 0; i < 4; ++i) {
    int c = t + i * 256;
    float x = X[row * H_ + c] + R[row * H_ + c];
    vals[i] = x;
    s += x;
  }
  red[t] = s;
  __syncthreads();
  for (int off = 128; off > 0; off >>= 1) {
    if (t < off) red[t] += red[t + off];
    __syncthreads();
  }
  float mean = red[0] * (1.0f / H_);
  __syncthreads();
  float vs = 0.f;
#pragma unroll
  for (int i = 0; i < 4; ++i) { float d = vals[i] - mean; vs += d * d; }
  red[t] = vs;
  __syncthreads();
  for (int off = 128; off > 0; off >>= 1) {
    if (t < off) red[t] += red[t + off];
    __syncthreads();
  }
  float inv = rsqrtf(red[0] * (1.0f / H_) + 1e-9f);
#pragma unroll
  for (int i = 0; i < 4; ++i) {
    int c = t + i * 256;
    Out[row * H_ + c] = (vals[i] - mean) * inv * gamma[c] + beta[c];
  }
}

extern "C" void kernel_launch(void* const* d_in, const int* in_sizes, int n_in,
                              void* d_out, int out_size, void* d_ws, size_t ws_size,
                              hipStream_t stream) {
  (void)in_sizes; (void)n_in; (void)out_size; (void)ws_size;
  const float* x     = (const float*)d_in[0];
  const float* mask  = (const float*)d_in[1];
  const float* w_in  = (const float*)d_in[2];
  const float* b_in  = (const float*)d_in[3];
  const float* wq    = (const float*)d_in[4];
  const float* bq    = (const float*)d_in[5];
  const float* wk    = (const float*)d_in[6];
  const float* bk    = (const float*)d_in[7];
  const float* wv    = (const float*)d_in[8];
  const float* bv    = (const float*)d_in[9];
  const float* wout  = (const float*)d_in[10];
  const float* bout  = (const float*)d_in[11];
  const float* gamma = (const float*)d_in[12];
  const float* beta  = (const float*)d_in[13];
  float* out = (float*)d_out;

  const size_t MH = (size_t)B_ * S_ * H_;  // 4M elements
  float* ws  = (float*)d_ws;
  float* h   = ws;            // input projection output (residual source)
  float* q   = ws + 1 * MH;
  float* k   = ws + 2 * MH;
  float* v   = ws + 3 * MH;
  float* ctx = ws + 4 * MH;
  float* ao  = q;             // attention-output GEMM reuses q buffer

  dim3 gg(8, 32), gb(256);                       // N/128 x M/128 tiles
  gemm_bias_kernel<<<gg, gb, 0, stream>>>(x, w_in, b_in, h);
  gemm_bias_kernel<<<gg, gb, 0, stream>>>(h, wq, bq, q);
  gemm_bias_kernel<<<gg, gb, 0, stream>>>(h, wk, bk, k);
  gemm_bias_kernel<<<gg, gb, 0, stream>>>(h, wv, bv, v);
  attn_kernel<<<dim3(S_ / 64, NH_, B_), 128, 0, stream>>>(q, k, v, mask, ctx);
  gemm_bias_kernel<<<gg, gb, 0, stream>>>(ctx, wout, bout, ao);
  ln_kernel<<<dim3(B_ * S_), 256, 0, stream>>>(ao, h, gamma, beta, out);
}
